// TransformerBlock_67233418051825
// MI455X (gfx1250) — matrix operand order repaired and checked by reading, not compiled, not run on hardware
//
#include <hip/hip_runtime.h>
#include <hip/hip_bf16.h>
#include <math.h>

#define D_MODEL 2048
#define N_HEADS 16
#define HEAD_DIM 128
#define SEQ 2048
#define BATCH 2
#define FF_DIM 8192
#define M_ROWS (BATCH * SEQ)
#define LN_EPS 1e-5f
#define NEG_BIG (-3.0e38f)

typedef __bf16 bf16;
typedef __bf16 v16bf __attribute__((ext_vector_type(16)));
typedef float  v8f   __attribute__((ext_vector_type(8)));

union FragBF { v16bf v; uint4 q[2]; };
union Pack8  { bf16 h[8]; uint4 q; };

static __device__ inline v8f vzero8() {
    v8f v;
#pragma unroll
    for (int i = 0; i < 8; ++i) v[i] = 0.0f;
    return v;
}

static __device__ inline v8f wmma_bf16(v16bf a, v16bf b, v8f c) {
    // D = A(16x32 bf16) * B(32x16 bf16) + C(16x16 f32)
    return __builtin_amdgcn_wmma_f32_16x16x32_bf16(false, a, false, b, (short)0, c,
                                                   false, false);
}

// Async memory->LDS copy of 2x16B (32 contiguous bytes), tracked by ASYNCcnt.
// INST_OFFSET is added to BOTH the LDS and the global address (ISA 08, 4.4).
static __device__ inline void async_copy_32B(unsigned lds_off, const bf16* gptr) {
    unsigned long long ga = (unsigned long long)(uintptr_t)gptr;
    asm volatile("global_load_async_to_lds_b128 %0, %1, off"
                 :: "v"(lds_off), "v"(ga) : "memory");
    asm volatile("global_load_async_to_lds_b128 %0, %1, off offset:16"
                 :: "v"(lds_off), "v"(ga) : "memory");
}

#define WAIT_ASYNC_0() asm volatile("s_wait_asynccnt 0x0" ::: "memory")
#define WAIT_ASYNC_4() asm volatile("s_wait_asynccnt 0x4" ::: "memory")

// ---------------------------------------------------------------------------
// fp32 -> bf16 cast (grid stride)
// ---------------------------------------------------------------------------
__global__ void cast_f32_to_bf16_kernel(const float* __restrict__ in,
                                        bf16* __restrict__ out, int n) {
    int i = blockIdx.x * blockDim.x + threadIdx.x;
    int stride = gridDim.x * blockDim.x;
    for (; i < n; i += stride) out[i] = (bf16)in[i];
}

// ---------------------------------------------------------------------------
// LayerNorm over D_MODEL, output bf16. One block (256 threads) per row.
// ---------------------------------------------------------------------------
__global__ __launch_bounds__(256) void ln_bf16_kernel(const float* __restrict__ x,
                                                      const float* __restrict__ w,
                                                      const float* __restrict__ bias,
                                                      bf16* __restrict__ out) {
    const int D = D_MODEL;
    int row = blockIdx.x;
    int tid = threadIdx.x;
    const float* xr = x + (size_t)row * D;
    int base = tid * 8;  // 256 * 8 = 2048
    float vals[8];
    float4 a = *(const float4*)(xr + base);
    float4 b4 = *(const float4*)(xr + base + 4);
    vals[0] = a.x; vals[1] = a.y; vals[2] = a.z; vals[3] = a.w;
    vals[4] = b4.x; vals[5] = b4.y; vals[6] = b4.z; vals[7] = b4.w;
    float lsum = 0.0f, lsq = 0.0f;
#pragma unroll
    for (int j = 0; j < 8; ++j) { lsum += vals[j]; lsq += vals[j] * vals[j]; }

    __shared__ float s1[256];
    __shared__ float s2[256];
    s1[tid] = lsum; s2[tid] = lsq;
    __syncthreads();
    for (int off = 128; off > 0; off >>= 1) {
        if (tid < off) { s1[tid] += s1[tid + off]; s2[tid] += s2[tid + off]; }
        __syncthreads();
    }
    float mu = s1[0] * (1.0f / D);
    float var = s2[0] * (1.0f / D) - mu * mu;
    float rs = rsqrtf(var + LN_EPS);

    bf16* orow = out + (size_t)row * D;
#pragma unroll
    for (int j = 0; j < 8; ++j) {
        int c = base + j;
        orow[c] = (bf16)((vals[j] - mu) * rs * w[c] + bias[c]);
    }
}

// ---------------------------------------------------------------------------
// Tiled WMMA GEMM: C[m,n] = sum_k A[m,k] * W[n,k]   (i.e. A @ W^T)
// A: [Mdim,Kdim] bf16 row-major, W: [Ndim,Kdim] bf16 row-major.
// Block tile 128x128, BK=32; 8 waves each computing 32x64 (2x4 WMMA tiles).
// K-tiles are staged into double-buffered LDS with GLOBAL_LOAD_ASYNC_TO_LDS
// (ASYNCcnt), overlapping the next tile's HBM/L2 latency with WMMA work.
// ---------------------------------------------------------------------------
enum { OUT_BF16 = 0, OUT_F32_RESID = 1, OUT_GELU_BF16 = 2, OUT_VT = 3 };

template <int MODE>
__global__ __launch_bounds__(256) void gemm_wmma_kernel(
    const bf16* __restrict__ A, const bf16* __restrict__ W,
    void* __restrict__ Cout, const float* __restrict__ resid,
    int Mdim, int Ndim, int Kdim) {
    // [buf][A=0/B=1][row][col], row stride 40 keeps 16B alignment (80B) + pad
    __shared__ bf16 sT[2][2][128][40];

    const int tid = threadIdx.x;
    const int wave = tid >> 5;
    const int lane = tid & 31;
    const int lhalf = lane >> 4;   // 0 or 1
    const int l16 = lane & 15;
    const int wm = wave >> 1;      // 0..3 -> rows of 32
    const int wn = wave & 1;       // 0..1 -> cols of 64
    const int gm0 = blockIdx.y * 128;
    const int gn0 = blockIdx.x * 128;

    const int ldRow = tid >> 1;          // 0..127
    const int ldCol = (tid & 1) * 16;    // 0 or 16

    // LDS byte offset of this thread's landing slot (low 32 bits of the
    // generic pointer to LDS are the in-allocation byte offset).
    const unsigned ldsBase = (unsigned)(uintptr_t)&sT[0][0][0][0];
    const unsigned slotOff = (unsigned)(ldRow * 40 + ldCol) * 2u;
    const unsigned bufStride = 2u * 128u * 40u * 2u;     // one [A+B] buffer
    const unsigned abStride  = 128u * 40u * 2u;          // A -> B within buffer
    const unsigned myA[2] = { ldsBase + slotOff, ldsBase + bufStride + slotOff };
    const unsigned myB[2] = { myA[0] + abStride, myA[1] + abStride };

    const bf16* gA = A + (size_t)(gm0 + ldRow) * Kdim + ldCol;
    const bf16* gB = W + (size_t)(gn0 + ldRow) * Kdim + ldCol;

    v8f acc[2][4];
#pragma unroll
    for (int i = 0; i < 2; ++i)
#pragma unroll
        for (int j = 0; j < 4; ++j) acc[i][j] = vzero8();

    const int nk = Kdim >> 5;  // K tiles of 32

    // Preload tile 0 into buffer 0.
    async_copy_32B(myA[0], gA);
    async_copy_32B(myB[0], gB);

    for (int t = 0; t < nk; ++t) {
        const int buf = t & 1;
        if (t + 1 < nk) {  // issue next tile into the other buffer
            const bf16* nA = gA + (size_t)(t + 1) * 32;
            const bf16* nB = gB + (size_t)(t + 1) * 32;
            async_copy_32B(myA[buf ^ 1], nA);
            async_copy_32B(myB[buf ^ 1], nB);
            WAIT_ASYNC_4();   // tile t landed; tile t+1 (4 ops) still in flight
        } else {
            WAIT_ASYNC_0();
        }
        __syncthreads();      // every wave's tile-t data is in LDS

        FragBF afr[2], bfr[4];
#pragma unroll
        for (int i = 0; i < 2; ++i) {
            int r = wm * 32 + i * 16 + l16;
            int c = lhalf * 8;  // lanes 0-15: K 0-7 & 16-23; lanes 16-31: 8-15 & 24-31
            afr[i].q[0] = *(const uint4*)&sT[buf][0][r][c];
            afr[i].q[1] = *(const uint4*)&sT[buf][0][r][c + 16];
        }
#pragma unroll
        for (int j = 0; j < 4; ++j) {
            int r = wn * 64 + j * 16 + l16;
            int c = lhalf * 8;  // lanes 0-15: K 0-7 & 16-23; lanes 16-31: 8-15 & 24-31
            bfr[j].q[0] = *(const uint4*)&sT[buf][1][r][c];
            bfr[j].q[1] = *(const uint4*)&sT[buf][1][r][c + 16];
        }
#pragma unroll
        for (int i = 0; i < 2; ++i)
#pragma unroll
            for (int j = 0; j < 4; ++j)
                acc[i][j] = wmma_bf16(afr[i].v, bfr[j].v, acc[i][j]);

        __syncthreads();  // all waves done reading buf before it is refilled
    }

    // Epilogue. C/D layout: vgpr r, lanes 0-15 => M=r; lanes 16-31 => M=8+r.
#pragma unroll
    for (int i = 0; i < 2; ++i) {
#pragma unroll
        for (int j = 0; j < 4; ++j) {
            int mbase = gm0 + wm * 32 + i * 16 + lhalf * 8;
            int n = gn0 + wn * 64 + j * 16 + l16;
            if (MODE == OUT_BF16) {
                bf16* out = (bf16*)Cout;
#pragma unroll
                for (int r = 0; r < 8; ++r)
                    out[(size_t)(mbase + r) * Ndim + n] = (bf16)acc[i][j][r];
            } else if (MODE == OUT_F32_RESID) {
                float* out = (float*)Cout;
#pragma unroll
                for (int r = 0; r < 8; ++r) {
                    size_t idx = (size_t)(mbase + r) * Ndim + n;
                    out[idx] = resid[idx] + acc[i][j][r];
                }
            } else if (MODE == OUT_GELU_BF16) {
                bf16* out = (bf16*)Cout;
#pragma unroll
                for (int r = 0; r < 8; ++r) {
                    float v = acc[i][j][r];
                    float g = 0.5f * v * (1.0f + erff(v * 0.70710678118654752f));
                    out[(size_t)(mbase + r) * Ndim + n] = (bf16)g;
                }
            } else {  // OUT_VT: write transposed [B, H, Hd, S], 8 s-values packed
                Pack8 pk;
#pragma unroll
                for (int r = 0; r < 8; ++r) pk.h[r] = (bf16)acc[i][j][r];
                int bb = mbase / SEQ;
                int s0 = mbase % SEQ;       // multiple of 8
                int hh = n / HEAD_DIM;
                int hd = n % HEAD_DIM;
                bf16* out = (bf16*)Cout;
                *(uint4*)(out + ((size_t)(bb * N_HEADS + hh) * HEAD_DIM + hd) * SEQ + s0) = pk.q;
            }
        }
    }
}

// ---------------------------------------------------------------------------
// Flash attention (causal). Block = 256 threads = 8 waves; wave handles 16
// query rows. Q A-frags preloaded; K B-frags from global; V read from the
// pre-transposed [B,H,Hd,S] buffer so its B-frags are contiguous; P converted
// to A-layout through a per-wave LDS slice (no block barriers needed).
// ---------------------------------------------------------------------------
__global__ __launch_bounds__(256) void attn_flash_kernel(
    const bf16* __restrict__ Q, const bf16* __restrict__ Kc,
    const bf16* __restrict__ VT, bf16* __restrict__ ctx) {
    const int S = SEQ, D = D_MODEL, Hd = HEAD_DIM, H = N_HEADS;
    __shared__ bf16 sP[8][16][40];

    const int tid = threadIdx.x;
    const int wave = tid >> 5;
    const int lane = tid & 31;
    const int lhalf = lane >> 4;
    const int l16 = lane & 15;
    const int bh = blockIdx.y;
    const int b = bh / H;
    const int h = bh % H;
    const int qt0 = blockIdx.x * 128 + wave * 16;
    const float scale = 0.08838834764831845f;  // 1/sqrt(128)

    const bf16* qbase = Q + (size_t)b * S * D + h * Hd;
    const bf16* kbase = Kc + (size_t)b * S * D + h * Hd;
    const bf16* vbase = VT + (size_t)(b * H + h) * Hd * S;

    FragBF qf[4];
#pragma unroll
    for (int kk = 0; kk < 4; ++kk) {
        const bf16* p = qbase + (size_t)(qt0 + l16) * D + kk * 32 + lhalf * 8;
        qf[kk].q[0] = *(const uint4*)p;
        qf[kk].q[1] = *(const uint4*)(p + 16);
    }

    v8f o[8];
#pragma unroll
    for (int dt = 0; dt < 8; ++dt) o[dt] = vzero8();
    float Mr[8], Lr[8];
#pragma unroll
    for (int r = 0; r < 8; ++r) { Mr[r] = NEG_BIG; Lr[r] = 0.0f; }

    for (int k0 = 0; k0 < qt0 + 16; k0 += 32) {
        // prefetch next key/value tiles into cache while we compute this one
        __builtin_prefetch(kbase + (size_t)(k0 + 32 + l16) * D, 0, 1);
        __builtin_prefetch(vbase + (size_t)(lane * 4) * S + k0 + 32, 0, 1);

        v8f sc[2];
        sc[0] = vzero8(); sc[1] = vzero8();
#pragma unroll
        for (int kk = 0; kk < 4; ++kk) {
#pragma unroll
            for (int t = 0; t < 2; ++t) {
                FragBF kf;
                const bf16* p =
                    kbase + (size_t)(k0 + t * 16 + l16) * D + kk * 32 + lhalf * 8;
                kf.q[0] = *(const uint4*)p;
                kf.q[1] = *(const uint4*)(p + 16);
                sc[t] = wmma_bf16(qf[kk].v, kf.v, sc[t]);
            }
        }

        float pv[2][8];
#pragma unroll
        for (int r = 0; r < 8; ++r) {
            int qrow = qt0 + r + lhalf * 8;
#pragma unroll
            for (int t = 0; t < 2; ++t) {
                int kidx = k0 + t * 16 + l16;
                float v = sc[t][r] * scale;
                if (kidx > qrow) v = NEG_BIG;
                pv[t][r] = v;
            }
            float rm = fmaxf(pv[0][r], pv[1][r]);
#pragma unroll
            for (int m = 1; m < 16; m <<= 1) rm = fmaxf(rm, __shfl_xor(rm, m, 32));
            float nm = fmaxf(Mr[r], rm);
            float alpha = __expf(Mr[r] - nm);
            Mr[r] = nm;
            float rs = 0.0f;
#pragma unroll
            for (int t = 0; t < 2; ++t) {
                pv[t][r] = __expf(pv[t][r] - nm);
                rs += pv[t][r];
            }
#pragma unroll
            for (int m = 1; m < 16; m <<= 1) rs += __shfl_xor(rs, m, 32);
            Lr[r] = Lr[r] * alpha + rs;
#pragma unroll
            for (int dt = 0; dt < 8; ++dt) o[dt][r] *= alpha;
        }

        // P: C-layout regs -> LDS (m,n) -> A-layout frag (same wave, DS in-order)
#pragma unroll
        for (int t = 0; t < 2; ++t)
#pragma unroll
            for (int r = 0; r < 8; ++r)
                sP[wave][r + lhalf * 8][t * 16 + l16] = (bf16)pv[t][r];

        FragBF pf;
        pf.q[0] = *(const uint4*)&sP[wave][l16][lhalf * 8];
        pf.q[1] = *(const uint4*)&sP[wave][l16][lhalf * 8 + 16];

#pragma unroll
        for (int dt = 0; dt < 8; ++dt) {
            FragBF vf;
            const bf16* p = vbase + (size_t)(dt * 16 + l16) * S + k0 + lhalf * 8;
            vf.q[0] = *(const uint4*)p;
            vf.q[1] = *(const uint4*)(p + 16);
            o[dt] = wmma_bf16(pf.v, vf.v, o[dt]);
        }
    }

#pragma unroll
    for (int r = 0; r < 8; ++r) {
        float inv = 1.0f / Lr[r];
        size_t rowoff = (size_t)(b * S + qt0 + r + lhalf * 8) * D + h * Hd;
#pragma unroll
        for (int dt = 0; dt < 8; ++dt)
            ctx[rowoff + dt * 16 + l16] = (bf16)(o[dt][r] * inv);
    }
}

// ---------------------------------------------------------------------------
// Host launcher
// ---------------------------------------------------------------------------
extern "C" void kernel_launch(void* const* d_in, const int* in_sizes, int n_in,
                              void* d_out, int out_size, void* d_ws, size_t ws_size,
                              hipStream_t stream) {
    const float* x       = (const float*)d_in[0];
    const float* tok     = (const float*)d_in[1];
    const float* Wq      = (const float*)d_in[2];
    const float* Wk      = (const float*)d_in[3];
    const float* Wv      = (const float*)d_in[4];
    const float* Wo      = (const float*)d_in[5];
    const float* W1      = (const float*)d_in[6];
    const float* W2      = (const float*)d_in[7];
    const float* ln1w    = (const float*)d_in[8];
    const float* ln1b    = (const float*)d_in[9];
    const float* ln2w    = (const float*)d_in[10];
    const float* ln2b    = (const float*)d_in[11];

    const int D = D_MODEL, M = M_ROWS, FF = FF_DIM;

    char* base = (char*)d_ws;
    size_t off = 0;
    auto carve = [&](size_t bytes) -> char* {
        char* p = base + off;
        off = (off + bytes + 255) & ~(size_t)255;
        return p;
    };
    bf16* wq_b  = (bf16*)carve((size_t)D * D * 2);
    bf16* wk_b  = (bf16*)carve((size_t)D * D * 2);
    bf16* wv_b  = (bf16*)carve((size_t)D * D * 2);
    bf16* wo_b  = (bf16*)carve((size_t)D * D * 2);
    bf16* w1_b  = (bf16*)carve((size_t)FF * D * 2);
    bf16* w2_b  = (bf16*)carve((size_t)D * FF * 2);
    bf16* hn_b  = (bf16*)carve((size_t)M * D * 2);   // LN1 out
    bf16* tok_b = (bf16*)carve((size_t)M * D * 2);   // token bf16, reused as LN2 out
    // q/k/vt/ctx region (4 * M*D*2 = M*FF*2) is reused for the FFN hidden acts
    char* big   = carve((size_t)M * FF * 2);
    bf16* q_b   = (bf16*)big;
    bf16* k_b   = (bf16*)(big + (size_t)M * D * 2);
    bf16* vt_b  = (bf16*)(big + (size_t)M * D * 4);
    bf16* ctx_b = (bf16*)(big + (size_t)M * D * 6);
    bf16* ffh_b = (bf16*)big;                        // aliases q/k/vt/ctx (dead by then)
    float* xmid = (float*)carve((size_t)M * D * 4);
    (void)ws_size; (void)in_sizes; (void)n_in; (void)out_size;

    const int CT = 256;
    auto cgrid = [&](size_t n) { return dim3((unsigned)(((n + CT - 1) / CT > 4096) ? 4096 : (n + CT - 1) / CT)); };

    // 1. weights -> bf16
    cast_f32_to_bf16_kernel<<<cgrid((size_t)D * D), CT, 0, stream>>>(Wq, wq_b, D * D);
    cast_f32_to_bf16_kernel<<<cgrid((size_t)D * D), CT, 0, stream>>>(Wk, wk_b, D * D);
    cast_f32_to_bf16_kernel<<<cgrid((size_t)D * D), CT, 0, stream>>>(Wv, wv_b, D * D);
    cast_f32_to_bf16_kernel<<<cgrid((size_t)D * D), CT, 0, stream>>>(Wo, wo_b, D * D);
    cast_f32_to_bf16_kernel<<<cgrid((size_t)FF * D), CT, 0, stream>>>(W1, w1_b, FF * D);
    cast_f32_to_bf16_kernel<<<cgrid((size_t)FF * D), CT, 0, stream>>>(W2, w2_b, FF * D);

    // 2. LN1 + token cast
    ln_bf16_kernel<<<M, 256, 0, stream>>>(x, ln1w, ln1b, hn_b);
    cast_f32_to_bf16_kernel<<<cgrid((size_t)M * D), CT, 0, stream>>>(tok, tok_b, M * D);

    // 3. Q, K projections (bf16 out), V projection (transposed bf16 out)
    dim3 gDD(D / 128, M / 128);
    gemm_wmma_kernel<OUT_BF16><<<gDD, 256, 0, stream>>>(hn_b, wq_b, q_b, nullptr, M, D, D);
    gemm_wmma_kernel<OUT_BF16><<<gDD, 256, 0, stream>>>(hn_b, wk_b, k_b, nullptr, M, D, D);
    gemm_wmma_kernel<OUT_VT><<<gDD, 256, 0, stream>>>(tok_b, wv_b, vt_b, nullptr, M, D, D);

    // 4. causal flash attention
    dim3 gAttn(SEQ / 128, BATCH * N_HEADS);
    attn_flash_kernel<<<gAttn, 256, 0, stream>>>(q_b, k_b, vt_b, ctx_b);

    // 5. output projection + residual -> xmid (fp32)
    gemm_wmma_kernel<OUT_F32_RESID><<<gDD, 256, 0, stream>>>(ctx_b, wo_b, xmid, x, M, D, D);

    // 6. LN2 (reuse tok_b), FFN1 with GELU, FFN2 with residual -> d_out
    ln_bf16_kernel<<<M, 256, 0, stream>>>(xmid, ln2w, ln2b, tok_b);
    dim3 gFF1(FF / 128, M / 128);
    gemm_wmma_kernel<OUT_GELU_BF16><<<gFF1, 256, 0, stream>>>(tok_b, w1_b, ffh_b, nullptr, M, FF, D);
    gemm_wmma_kernel<OUT_F32_RESID><<<gDD, 256, 0, stream>>>(ffh_b, w2_b, (float*)d_out, xmid, M, D, FF);
}
